// MultiHeadAttention_20194936226178
// MI455X (gfx1250) — matrix operand order repaired and checked by reading, not compiled, not run on hardware
//
#include <hip/hip_runtime.h>
#include <hip/hip_bf16.h>

typedef _Float16 f16;
typedef __attribute__((ext_vector_type(16))) _Float16 v16h;
typedef __attribute__((ext_vector_type(8)))  float    v8f;
typedef __attribute__((ext_vector_type(4)))  int      v4i;

#define GLOBAL_AS __attribute__((address_space(1)))
#define LDS_AS    __attribute__((address_space(3)))

namespace {
constexpr int kB  = 4;
constexpr int kS  = 2048;
constexpr int kD  = 512;
constexpr int kH  = 8;
constexpr int kHd = 64;
constexpr float kScale = 0.125f;   // 1/sqrt(64)
}

// ---------------- WMMA helper (CDNA5 16x16x32 f16, f32 accum) ----------------

static __device__ inline v8f wmma32(v16h a, v16h b, v8f c) {
  // 8 args: (neg_a, A, neg_b, B, c_mod, C, reuse_a, reuse_b)
  return __builtin_amdgcn_wmma_f32_16x16x32_f16(false, a, false, b, (short)0, c,
                                                false, false);
}

// ---------------- CDNA5 async global->LDS copy (ASYNCcnt path) ----------------

static __device__ inline void async_copy_b128(const f16* g, f16* l) {
#if __has_builtin(__builtin_amdgcn_global_load_async_to_lds_b128)
  __builtin_amdgcn_global_load_async_to_lds_b128(
      (GLOBAL_AS v4i*)g, (LDS_AS v4i*)l, 0, 0);
#else
  asm volatile("global_load_async_to_lds_b128 %0, %1, off" ::"v"(
                   (unsigned)(unsigned long long)l),
               "v"(g)
               : "memory");
#endif
}

static __device__ inline void wait_async0() {
#if __has_builtin(__builtin_amdgcn_s_wait_asynccnt)
  __builtin_amdgcn_s_wait_asynccnt(0);
#else
  asm volatile("s_wait_asynccnt 0x0" ::: "memory");
#endif
}

// ---------------- Fragment loaders ----------------

// A-fragment: 16x32 (MxK) f16. ISA 7.12.2: lane row = lane&15; VGPR v holds
// K = (v>>2)*16 + (lane>=16 ? 8 : 0) + (v&3)*2 (+1).
static __device__ inline v16h load_a_h(const f16* __restrict__ base, int ld) {
  const int lane = threadIdx.x & 31;
  const int row  = lane & 15;
  const int half = lane >> 4;
  const f16* p = base + (size_t)row * ld;
  v16h a;
#pragma unroll
  for (int i = 0; i < 8; ++i) {
    const int k = ((i >> 2) << 4) + (half << 3) + ((i & 3) << 1);
    a[2 * i]     = p[k];
    a[2 * i + 1] = p[k + 1];
  }
  return a;
}

// Same A layout, source f32, convert to f16 in-register.
static __device__ inline v16h load_a_f(const float* __restrict__ base, int ld) {
  const int lane = threadIdx.x & 31;
  const int row  = lane & 15;
  const int half = lane >> 4;
  const float* p = base + (size_t)row * ld;
  v16h a;
#pragma unroll
  for (int i = 0; i < 8; ++i) {
    const int k = ((i >> 2) << 4) + (half << 3) + ((i & 3) << 1);
    a[2 * i]     = (f16)p[k];
    a[2 * i + 1] = (f16)p[k + 1];
  }
  return a;
}

// B-fragment: 32x16 (KxN) built from an N-major matrix M[n][k] (A@M^T form).
// Same K order as A: lane col = lane&15; VGPR v holds K = (v>>2)*16 + half*8 + (v&3)*2.
static __device__ inline v16h load_bT_h(const f16* __restrict__ base, int ld) {
  const int lane = threadIdx.x & 31;
  const int col  = lane & 15;
  const int half = lane >> 4;
  const f16* p = base + (size_t)col * ld;
  v16h b;
#pragma unroll
  for (int i = 0; i < 8; ++i) {
    const int k = ((i >> 2) << 4) + (half << 3) + ((i & 3) << 1);
    b[2 * i]     = p[k];
    b[2 * i + 1] = p[k + 1];
  }
  return b;
}

static __device__ inline v16h load_bT_f(const float* __restrict__ base, int ld) {
  const int lane = threadIdx.x & 31;
  const int col  = lane & 15;
  const int half = lane >> 4;
  const float* p = base + (size_t)col * ld;
  v16h b;
#pragma unroll
  for (int i = 0; i < 8; ++i) {
    const int k = ((i >> 2) << 4) + (half << 3) + ((i & 3) << 1);
    b[2 * i]     = (f16)p[k];
    b[2 * i + 1] = (f16)p[k + 1];
  }
  return b;
}

// Wave32 half-group (16-lane) reductions matching C-fragment column striping.
// Only used ONCE per row now (lane-local online softmax in the hot loop).
static __device__ inline float hg_max(float x) {
  x = fmaxf(x, __shfl_xor(x, 1));
  x = fmaxf(x, __shfl_xor(x, 2));
  x = fmaxf(x, __shfl_xor(x, 4));
  x = fmaxf(x, __shfl_xor(x, 8));
  return x;
}
static __device__ inline float hg_sum(float x) {
  x += __shfl_xor(x, 1);
  x += __shfl_xor(x, 2);
  x += __shfl_xor(x, 4);
  x += __shfl_xor(x, 8);
  return x;
}

// ---------------- Kernel 1: q/k/v projection GEMM ----------------
// C[m][n] = sum_k X[m][k] * W[n][k].
// TRANSPOSED=false: store [b][h][s][d] (q, k). TRANSPOSED=true: [b][h][d][s] (v),
// so the attn@V B-operand becomes a contiguous bT load later.
template <bool TRANSPOSED>
__global__ __launch_bounds__(128) void proj_qkv_kernel(
    const float* __restrict__ X, const float* __restrict__ W,
    f16* __restrict__ out) {
  const int wave = threadIdx.x >> 5;
  const int lane = threadIdx.x & 31;
  const int m0 = blockIdx.x * 64 + (wave >> 1) * 32;
  const int n0 = blockIdx.y * 64 + (wave & 1) * 32;

  v8f zero = {};
  v8f c[2][2] = {{zero, zero}, {zero, zero}};

  for (int kk = 0; kk < kD; kk += 32) {
    v16h a0 = load_a_f(X + (size_t)m0 * kD + kk, kD);
    v16h a1 = load_a_f(X + (size_t)(m0 + 16) * kD + kk, kD);
    v16h b0 = load_bT_f(W + (size_t)n0 * kD + kk, kD);
    v16h b1 = load_bT_f(W + (size_t)(n0 + 16) * kD + kk, kD);
    c[0][0] = wmma32(a0, b0, c[0][0]);
    c[0][1] = wmma32(a0, b1, c[0][1]);
    c[1][0] = wmma32(a1, b0, c[1][0]);
    c[1][1] = wmma32(a1, b1, c[1][1]);
  }

  const int half = lane >> 4;
  const int cn   = lane & 15;
#pragma unroll
  for (int i = 0; i < 2; ++i)
#pragma unroll
    for (int j = 0; j < 2; ++j)
#pragma unroll
      for (int r = 0; r < 8; ++r) {
        const int m = m0 + i * 16 + r + half * 8;
        const int n = n0 + j * 16 + cn;
        const int b = m >> 11, s = m & (kS - 1);
        const int h = n >> 6,  d = n & (kHd - 1);
        if (TRANSPOSED)
          out[(((size_t)b * kH + h) * kHd + d) * kS + s] = (f16)c[i][j][r];
        else
          out[((((size_t)b * kH + h) * kS + s) << 6) + d] = (f16)c[i][j][r];
      }
}

// ---------------- Kernel 2: softmax stats (pass 1, no score store) ----------------
// 4 waves share one (b,h): K tiles staged once per block via double-buffered
// async global->LDS copies. Softmax statistics are kept LANE-LOCAL in the hot
// loop (each lane owns a fixed subset of columns); the 16-lane merge happens
// once per row at the end.
__global__ __launch_bounds__(128) void attn_stats_kernel(
    const f16* __restrict__ q, const f16* __restrict__ k,
    float* __restrict__ row_m, float* __restrict__ row_l) {
  __shared__ f16 ldsK[2][32 * 72];   // 32 keys x 64 hd, padded stride 72

  const int tid  = threadIdx.x;
  const int bh   = blockIdx.x;
  const int wave = tid >> 5;
  const int lane = tid & 31;
  const int m0   = blockIdx.y * 64 + wave * 16;

  const f16* qb = q + ((size_t)bh * kS + m0) * kHd;
  const f16* kb = k + (size_t)bh * kS * kHd;

  v16h a0 = load_a_h(qb, kHd);       // K = 0..31 of hd
  v16h a1 = load_a_h(qb + 32, kHd);  // K = 32..63

  float rm[8], rl[8];
#pragma unroll
  for (int r = 0; r < 8; ++r) { rm[r] = -1e30f; rl[r] = 0.0f; }

  const int rk = tid >> 3, ck = (tid & 7) * 8;  // per-lane staging coords
  auto stageK = [&](int buf, int n0) {
    async_copy_b128(kb + (size_t)(n0 + rk) * kHd + ck, &ldsK[buf][rk * 72 + ck]);
    async_copy_b128(kb + (size_t)(n0 + rk + 16) * kHd + ck,
                    &ldsK[buf][(rk + 16) * 72 + ck]);
  };

  stageK(0, 0);
  constexpr int kIters = kS / 32;   // 64
  for (int it = 0; it < kIters; ++it) {
    const int buf = it & 1;
    wait_async0();
    __syncthreads();
    if (it + 1 < kIters) {
      stageK(buf ^ 1, (it + 1) * 32);             // overlap DMA with WMMA
      __builtin_prefetch(kb + (size_t)(it + 4) * 32 * kHd, 0, 1);  // L2 lookahead
    }
#pragma unroll
    for (int t = 0; t < 2; ++t) {
      v16h b0 = load_bT_h(&ldsK[buf][t * 16 * 72], 72);
      v16h b1 = load_bT_h(&ldsK[buf][t * 16 * 72 + 32], 72);
      v8f c = {};
      c = wmma32(a0, b0, c);
      c = wmma32(a1, b1, c);
#pragma unroll
      for (int r = 0; r < 8; ++r) {   // lane-local online (max, sumexp)
        const float sv = c[r] * kScale;
        const float mn = fmaxf(rm[r], sv);
        rl[r] = rl[r] * __expf(rm[r] - mn) + __expf(sv - mn);
        rm[r] = mn;
      }
    }
    __syncthreads();
  }

  // One cross-lane merge per row: M = max_lane rm, L = sum_lane rl*exp(rm-M).
  const int half = lane >> 4;
#pragma unroll
  for (int r = 0; r < 8; ++r) {
    const float M = hg_max(rm[r]);
    const float L = hg_sum(rl[r] * __expf(rm[r] - M));
    rm[r] = M;
    rl[r] = L;
  }
  if ((lane & 15) == 0) {
#pragma unroll
    for (int r = 0; r < 8; ++r) {
      const int row = m0 + r + half * 8;
      row_m[(size_t)bh * kS + row] = rm[r];
      row_l[(size_t)bh * kS + row] = rl[r];
    }
  }
}

// ---------------- Kernel 3: fused attn emit + ctx = P @ V (pass 2) ----------------
// Double-buffered async staging of K and V^T tiles (once per block); each wave
// recomputes its score tile from LDS, normalizes with pass-1 stats, streams
// attn out (the single mandatory 536 MB write), and feeds the probability tile
// through a per-wave LDS buffer (C-layout -> A-layout) into the attn@V WMMAs.
__global__ __launch_bounds__(128) void attn_emit_kernel(
    const f16* __restrict__ q, const f16* __restrict__ k,
    const f16* __restrict__ vt, const float* __restrict__ row_m,
    const float* __restrict__ row_l, float* __restrict__ attn,
    f16* __restrict__ ctx) {
  __shared__ f16 ldsK[2][32 * 72];   // 32 keys x 64 hd, padded stride 72
  __shared__ f16 ldsV[2][64 * 40];   // 64 hd rows x 32 keys, padded stride 40
  __shared__ f16 lp[4][16 * 40];     // per-wave 16x32 prob tile, stride 40

  const int tid  = threadIdx.x;
  const int bh   = blockIdx.x;
  const int b    = bh >> 3;
  const int h    = bh & (kH - 1);
  const int wave = tid >> 5;
  const int lane = tid & 31;
  const int m0   = blockIdx.y * 64 + wave * 16;
  const int half = lane >> 4;
  const int cn   = lane & 15;

  const f16* qb  = q + ((size_t)bh * kS + m0) * kHd;
  const f16* kb  = k + (size_t)bh * kS * kHd;
  const f16* vtb = vt + (size_t)bh * kHd * kS;   // [d][s]

  v16h a0 = load_a_h(qb, kHd);
  v16h a1 = load_a_h(qb + 32, kHd);

  float rm_[8], ri_[8];
#pragma unroll
  for (int r = 0; r < 8; ++r) {
    const int row = m0 + r + half * 8;
    rm_[r] = row_m[(size_t)bh * kS + row];
    ri_[r] = 1.0f / row_l[(size_t)bh * kS + row];
  }

  v8f zero = {};
  v8f acc[4] = {zero, zero, zero, zero};  // 16 rows x 64 hd cols
  float* ab = attn + ((size_t)bh * kS + m0) * kS;
  f16* myp = &lp[wave][0];

  const int rk = tid >> 3, ck = (tid & 7) * 8;  // K tile: row, 16B chunk
  const int rv = tid >> 2, cv = (tid & 3) * 8;  // V tile: row, 16B chunk
  auto stage = [&](int buf, int n0) {
    async_copy_b128(kb + (size_t)(n0 + rk) * kHd + ck, &ldsK[buf][rk * 72 + ck]);
    async_copy_b128(kb + (size_t)(n0 + rk + 16) * kHd + ck,
                    &ldsK[buf][(rk + 16) * 72 + ck]);
    async_copy_b128(vtb + (size_t)rv * kS + n0 + cv, &ldsV[buf][rv * 40 + cv]);
    async_copy_b128(vtb + (size_t)(rv + 32) * kS + n0 + cv,
                    &ldsV[buf][(rv + 32) * 40 + cv]);
  };

  stage(0, 0);
  constexpr int kIters = kS / 32;   // 64
  for (int it = 0; it < kIters; ++it) {
    const int n0  = it * 32;
    const int buf = it & 1;
    wait_async0();
    __syncthreads();
    if (it + 1 < kIters) stage(buf ^ 1, n0 + 32);  // overlap DMA with compute

    // ---- score tiles from LDS, normalize, emit attn, stage P ----
#pragma unroll
    for (int t = 0; t < 2; ++t) {
      v16h b0 = load_bT_h(&ldsK[buf][t * 16 * 72], 72);
      v16h b1 = load_bT_h(&ldsK[buf][t * 16 * 72 + 32], 72);
      v8f c = {};
      c = wmma32(a0, b0, c);
      c = wmma32(a1, b1, c);
      const int kt = n0 + t * 16;
#pragma unroll
      for (int r = 0; r < 8; ++r) {
        const float p = __expf(c[r] * kScale - rm_[r]) * ri_[r];
        ab[(size_t)(r + half * 8) * kS + kt + cn] = p;   // stream attn out
        myp[(r + half * 8) * 40 + t * 16 + cn] = (f16)p; // stage for A-frag
      }
    }
    // Same-wave DS ops are in-order (ISA ch.11); reload in A-fragment layout.
    v16h pa = load_a_h(myp, 40);
#pragma unroll
    for (int nt = 0; nt < 4; ++nt) {
      v16h bv = load_bT_h(&ldsV[buf][nt * 16 * 40], 40);
      acc[nt] = wmma32(pa, bv, acc[nt]);
    }
    __syncthreads();  // all waves done with buf before it is restaged
  }

  // Store ctx in concat layout [b][s][h*64+d] (f16 for the output GEMM).
#pragma unroll
  for (int nt = 0; nt < 4; ++nt)
#pragma unroll
    for (int r = 0; r < 8; ++r) {
      const int s = m0 + r + half * 8;
      const int d = h * kHd + nt * 16 + cn;
      ctx[((size_t)b * kS + s) * kD + d] = (f16)acc[nt][r];
    }
}

// ---------------- Kernel 4: output projection out = ctx @ Wo^T + bo ----------------
__global__ __launch_bounds__(128) void out_proj_kernel(
    const f16* __restrict__ ctx, const float* __restrict__ Wo,
    const float* __restrict__ bo, float* __restrict__ out) {
  const int wave = threadIdx.x >> 5;
  const int lane = threadIdx.x & 31;
  const int m0 = blockIdx.x * 64 + (wave >> 1) * 32;
  const int n0 = blockIdx.y * 64 + (wave & 1) * 32;

  v8f zero = {};
  v8f c[2][2] = {{zero, zero}, {zero, zero}};

  for (int kk = 0; kk < kD; kk += 32) {
    v16h a0 = load_a_h(ctx + (size_t)m0 * kD + kk, kD);
    v16h a1 = load_a_h(ctx + (size_t)(m0 + 16) * kD + kk, kD);
    v16h b0 = load_bT_f(Wo + (size_t)n0 * kD + kk, kD);
    v16h b1 = load_bT_f(Wo + (size_t)(n0 + 16) * kD + kk, kD);
    c[0][0] = wmma32(a0, b0, c[0][0]);
    c[0][1] = wmma32(a0, b1, c[0][1]);
    c[1][0] = wmma32(a1, b0, c[1][0]);
    c[1][1] = wmma32(a1, b1, c[1][1]);
  }

  const int half = lane >> 4;
  const int cn   = lane & 15;
#pragma unroll
  for (int i = 0; i < 2; ++i)
#pragma unroll
    for (int j = 0; j < 2; ++j) {
      const int n = n0 + j * 16 + cn;
      const float bias = bo[n];
#pragma unroll
      for (int r = 0; r < 8; ++r) {
        const int m = m0 + i * 16 + r + half * 8;
        out[(size_t)m * kD + n] = c[i][j][r] + bias;
      }
    }
}

// ---------------- Host-side launcher ----------------
extern "C" void kernel_launch(void* const* d_in, const int* in_sizes, int n_in,
                              void* d_out, int out_size, void* d_ws,
                              size_t ws_size, hipStream_t stream) {
  (void)in_sizes; (void)n_in; (void)out_size; (void)ws_size;

  const float* query = (const float*)d_in[0];
  const float* key_  = (const float*)d_in[1];
  const float* value = (const float*)d_in[2];
  const float* Wq    = (const float*)d_in[3];
  const float* Wk    = (const float*)d_in[4];
  const float* Wv    = (const float*)d_in[5];
  const float* Wo    = (const float*)d_in[6];
  const float* bo    = (const float*)d_in[7];

  // Workspace carve-out (needs ~33 MB):
  //   q/k f16 head-major [B,H,S,hd], v^T f16 [B,H,hd,S] : 3 x 8 MB
  //   ctx f16 concat [B,S,D]                            : 8 MB
  //   row max / row sumexp f32 [B*H*S]                  : 2 x 256 KB
  const size_t nQKV = (size_t)kB * kH * kS * kHd;  // == B*S*D elements
  f16* qf = (f16*)d_ws;
  f16* kf = qf + nQKV;
  f16* vf = kf + nQKV;   // transposed layout [b][h][d][s]
  f16* cf = vf + nQKV;
  float* rowm = (float*)(cf + nQKV);
  float* rowl = rowm + (size_t)kB * kH * kS;

  float* out  = (float*)d_out;                      // [B,S,D]
  float* attn = out + (size_t)kB * kS * kD;         // [B,H,S,S]

  const dim3 blk(128);
  const dim3 gproj(kB * kS / 64, kD / 64);          // (128, 8)
  const dim3 gattn(kB * kH, kS / 64);               // (32, 32)

  proj_qkv_kernel<false><<<gproj, blk, 0, stream>>>(query, Wq, qf);
  proj_qkv_kernel<false><<<gproj, blk, 0, stream>>>(key_,  Wk, kf);
  proj_qkv_kernel<true ><<<gproj, blk, 0, stream>>>(value, Wv, vf);
  attn_stats_kernel<<<gattn, blk, 0, stream>>>(qf, kf, rowm, rowl);
  attn_emit_kernel<<<gattn, blk, 0, stream>>>(qf, kf, vf, rowm, rowl, attn, cf);
  out_proj_kernel<<<gproj, blk, 0, stream>>>(cf, Wo, bo, out);
}
